// mLSTMblock_54872502174276
// MI455X (gfx1250) — matrix operand order repaired and checked by reading, not compiled, not run on hardware
//
#include <hip/hip_runtime.h>

constexpr int kB = 8;
constexpr int kT = 2048;
constexpr int kD = 128;
constexpr int kL = 32;              // chunk length (== one WMMA K-step for f16)
constexpr int kNC = kT / kL;        // 64 chunks
constexpr int kFragH = 512;         // halves per fragment (32 lanes x 16)
constexpr int kChunkH = 32 * kFragH; // halves per (batch,chunk) fragment block

typedef __attribute__((ext_vector_type(16))) _Float16 v16h;
typedef __attribute__((ext_vector_type(8)))  float    v8f;

struct __align__(32) SmemBlk { char data[32]; };

static __device__ __forceinline__ v8f wmma_f16(v16h a, v16h b, v8f c) {
  // D = A(16x32 f16) * B(32x16 f16) + C(16x16 f32)
  return __builtin_amdgcn_wmma_f32_16x16x32_f16(false, a, false, b, (short)0, c,
                                                false, false);
}

// Fragment resident in LDS in register layout: lane reads 16 contiguous halves
// (32 bytes -> two ds_load_b128).
static __device__ __forceinline__ v16h frag_ld(const _Float16* fr, int lane) {
  return *(const v16h*)(fr + lane * 16);
}

// Issue a flat 32KB global->LDS copy: 8 x global_load_async_to_lds_b128 per
// thread across 256 threads (ASYNCcnt-tracked; complete with s_wait_asynccnt).
static __device__ __forceinline__ void async_copy_32k(unsigned ldsbase,
                                                      const void* src, int tid) {
#pragma unroll
  for (int u = 0; u < 8; ++u) {
    unsigned boff = (unsigned)(tid * 16 + u * 4096);
    unsigned laddr = ldsbase + boff;
    asm volatile("global_load_async_to_lds_b128 %0, %1, %2"
                 :: "v"(laddr), "v"(boff), "s"(src) : "memory");
  }
}

// A-layout (16x32 MxK): lane l (m16=l&15, hi=l>>4), elem e -> k=e+16*hi
// B-layout (32x16 KxN): lane l: n=l&15, k = e + 16*(l>>4)
// C/D f32:              lane l: n=l&15, vgpr r -> m = r + 8*(l>>4)

// (row in [0,32), col in [0,32)) -> A-frag linear index, frag per 16 rows.
static __device__ __forceinline__ int a_scatter_idx(int row, int j) {
  int hib = (j >> 4) & 1;
  int e = j & 15;
  return (row >> 4) * kFragH + ((row & 15) + 16 * hib) * 16 + e;
}
// (row in [0,32), col in [0,128)) -> Q A-frag index, frag = (row>>4)*4 + col>>5
static __device__ __forceinline__ int a_frag_idx(int row, int col) {
  int dd = col & 31;
  int hib = (dd >> 4) & 1;
  int e = dd & 15;
  return ((row >> 4) * 4 + (col >> 5)) * kFragH + ((row & 15) + 16 * hib) * 16 + e;
}
// (k-row j in [0,32), col in [0,128)) -> B-frag linear index, frag per 16 cols.
static __device__ __forceinline__ int b_frag_idx(int j, int col) {
  return (col >> 4) * kFragH + ((col & 15) + 16 * (j >> 4)) * 16 + (j & 15);
}
// (n-col j in [0,32), k-row d in [0,128)) -> K^T frag index, frag = (d>>5)*2 + j>>4
static __device__ __forceinline__ int bt_frag_idx(int j, int d) {
  return ((d >> 5) * 2 + (j >> 4)) * kFragH + ((j & 15) + 16 * ((d >> 4) & 1)) * 16 + (d & 15);
}
// State element (i in [0,128), d in [0,128)) -> C0F (QC B-layout) index.
static __device__ __forceinline__ int c0f_idx(int i, int d) {
  return ((d >> 5) * 8 + (i >> 4)) * kFragH + ((i & 15) + 16 * ((d >> 4) & 1)) * 16 + (d & 15);
}

// ---------------------------------------------------------------------------
// Projections. q/k/v are written as f16 directly in WMMA fragment layout,
// one contiguous 32KB block per (batch, chunk):
//   [ QA 8 frags | KT 8 frags | KB 8 frags | VB 8 frags ]  (halves)
// o = sigmoid(xWo+bo) stays fp32 row-major for the epilogue.
// ---------------------------------------------------------------------------
__global__ void __launch_bounds__(256)
mlstm_proj_kernel(const float* __restrict__ x,
                  const float* __restrict__ Wq, const float* __restrict__ bq,
                  const float* __restrict__ Wk, const float* __restrict__ bk,
                  const float* __restrict__ Wv, const float* __restrict__ bv,
                  const float* __restrict__ Wo, const float* __restrict__ bo,
                  _Float16* __restrict__ frag, float* __restrict__ o) {
  const int row0 = blockIdx.x * 16;
  const int wave = threadIdx.x >> 5, lane = threadIdx.x & 31;
  const int m16 = lane & 15, hi = lane >> 4;

  // All 16 rows of this tile live in one chunk (chunk = 32 rows, row0 % 16 == 0)
  const int bb2 = row0 >> 11;                 // batch
  const int chx = (row0 & (kT - 1)) >> 5;     // chunk
  _Float16* fbase = frag + ((size_t)(bb2 * kNC + chx) * kChunkH);

  v16h afr[4];
#pragma unroll
  for (int kk = 0; kk < 4; ++kk) {
#pragma unroll
    for (int e = 0; e < 16; ++e) {
      int kq = kk * 32 + e + hi * 16;
      afr[kk][e] = (_Float16)x[(size_t)(row0 + m16) * kD + kq];
    }
  }

  for (int t = wave; t < 32; t += 8) {
    const int mat = t >> 3;
    const int n0 = (t & 7) * 16;
    const float* W  = mat == 0 ? Wq : mat == 1 ? Wk : mat == 2 ? Wv : Wo;
    const float* bb = mat == 0 ? bq : mat == 1 ? bk : mat == 2 ? bv : bo;

    v8f acc = {};
#pragma unroll
    for (int kk = 0; kk < 4; ++kk) {
      v16h bfr;
#pragma unroll
      for (int e = 0; e < 16; ++e) {
        int kq = kk * 32 + e + hi * 16;
        bfr[e] = (_Float16)W[kq * kD + n0 + m16];
      }
      acc = wmma_f16(afr[kk], bfr, acc);
    }
    const int nc = n0 + m16;
    const float bias = bb[nc];
#pragma unroll
    for (int r = 0; r < 8; ++r) {
      const int rg = row0 + r + hi * 8;        // global row
      const int s = rg & (kL - 1);             // row within chunk
      float val = acc[r] + bias;
      if (mat == 0) {
        fbase[a_frag_idx(s, nc)] = (_Float16)val;
      } else if (mat == 1) {
        val *= 0.088388347648318447f;          // 1/sqrt(128)
        fbase[8 * kFragH + bt_frag_idx(s, nc)] = (_Float16)val;
        fbase[16 * kFragH + b_frag_idx(s, nc)] = (_Float16)val;
      } else if (mat == 2) {
        fbase[24 * kFragH + b_frag_idx(s, nc)] = (_Float16)val;
      } else {
        val = 1.0f / (1.0f + __expf(-val));    // sigmoid
        o[(size_t)rg * kD + nc] = val;
      }
    }
  }
}

// ---------------------------------------------------------------------------
// Scalar gate projections: i_bar = x@Wi + bi, f_bar = x@Wf + bf  (D -> 1)
// ---------------------------------------------------------------------------
__global__ void __launch_bounds__(256)
mlstm_gate_kernel(const float* __restrict__ x,
                  const float* __restrict__ Wi, const float* __restrict__ bi,
                  const float* __restrict__ Wf, const float* __restrict__ bf,
                  float* __restrict__ ib, float* __restrict__ fb, int rows) {
  int r = blockIdx.x * blockDim.x + threadIdx.x;
  if (r >= rows) return;
  float si = 0.f, sf = 0.f;
#pragma unroll 4
  for (int d = 0; d < kD; ++d) {
    float xv = x[(size_t)r * kD + d];
    si += xv * Wi[d];
    sf += xv * Wf[d];
  }
  ib[r] = si + bi[0];
  fb[r] = sf + bf[0];
}

// ---------------------------------------------------------------------------
// Chunkwise-parallel mLSTM scan. One block (8 wave32s) per batch element.
// Double-buffered, software-pipelined async global->LDS fragment copies.
// ---------------------------------------------------------------------------
__global__ void __launch_bounds__(256)
mlstm_scan_kernel(const _Float16* __restrict__ frag, const float* __restrict__ og,
                  const float* __restrict__ ibg, const float* __restrict__ fbg,
                  float* __restrict__ out) {
  extern __shared__ SmemBlk smem_dyn[];
  // ---- f16 fragment region first (all offsets 1KB multiples => 32B aligned)
  _Float16* C0F  = (_Float16*)smem_dyn;       // 32 frags: f16 state, QC B-layout
  _Float16* qkv0 = C0F + 32 * kFragH;         // 32 frags: QKV buffer 0
  _Float16* qkv1 = qkv0 + 32 * kFragH;        // 32 frags: QKV buffer 1
  _Float16* VA   = qkv1 + 32 * kFragH;        // 8 frags: (w_L o V)^T A-tiles
  _Float16* SsA  = VA + 8 * kFragH;           // 2 frags: Shat A-tiles
  _Float16* WA   = SsA + 2 * kFragH;          // 2 frags: W A-tiles
  // ---- fp32 region ----
  float* C0   = (float*)(WA + 2 * kFragH);    // D*D   fp32 state
  float* SVb  = C0 + kD * kD;                 // L*D
  float* QCb  = SVb + kL * kD;                // L*D
  float* NMb  = QCb + kL * kD;                // L*D
  float* Wm   = NMb + kL * kD;                // L*L
  float* n0   = Wm + kL * kL;                 // D
  float* nnew = n0 + kD;                      // D
  float* Fc   = nnew + kD;                    // L
  float* msv  = Fc + kL;                      // L
  float* dec  = msv + kL;                     // L
  float* den  = dec + kL;                     // L
  float* ibs  = den + kL;                     // L
  float* fbs  = ibs + kL;                     // L
  float* mcar = fbs + kL;                     // 1

  const int b = blockIdx.x;
  const int tid = threadIdx.x;
  const int wave = tid >> 5, lane = tid & 31;
  const int m16 = lane & 15, hi = lane >> 4;

  // LDS byte addresses of the two QKV fragment buffers.
  const unsigned ldsBuf0 = __builtin_amdgcn_groupstaticsize() +
                           (unsigned)((char*)qkv0 - (char*)smem_dyn);
  const unsigned ldsBuf1 = ldsBuf0 + 32u * kFragH * (unsigned)sizeof(_Float16);

  // Prologue: start chunk 0's copy; it overlaps with the state zero-init.
  async_copy_32k(ldsBuf0, (const void*)(frag + (size_t)(b * kNC) * kChunkH), tid);

  for (int i = tid; i < kD * kD; i += 256) { C0[i] = 0.f; C0F[i] = (_Float16)0.f; }
  if (tid < kD) n0[tid] = 0.f;
  if (tid == 0) mcar[0] = 0.f;
  __syncthreads();

  float* outH     = out;                              // [B,T,D]
  float* outHlast = out + (size_t)kB * kT * kD;       // [B,D]
  float* outC     = outHlast + kB * kD;               // [B,D,D]
  float* outN     = outC + (size_t)kB * kD * kD;      // [B,D]
  float* outM     = outN + kB * kD;                   // [B,D]

  for (int ch = 0; ch < kNC; ++ch) {
    const int t0 = ch * kL;
    const size_t rowbase = (size_t)(b * kT + t0) * kD;
    const int cur = ch & 1;
    _Float16* QsA = cur ? qkv1 : qkv0;
    _Float16* KsT = QsA + 8 * kFragH;
    _Float16* KsB = KsT + 8 * kFragH;
    _Float16* VsB = KsB + 8 * kFragH;

    // Complete the copy issued one chunk ago (usually already done).
    asm volatile("s_wait_asynccnt 0" ::: "memory");
    if (tid < kL) {
      ibs[tid] = ibg[b * kT + t0 + tid];
      fbs[tid] = fbg[b * kT + t0 + tid];
    }
    // prefetch next chunk's output-gate rows (speculative, L2)
    if (ch + 1 < kNC)
      __builtin_prefetch((const char*)(og + rowbase + kL * kD) + tid * 64, 0, 1);
    __syncthreads();

    // Pipeline: kick off next chunk's fragment copy into the other buffer.
    if (ch + 1 < kNC)
      async_copy_32k(cur ? ldsBuf0 : ldsBuf1,
                     (const void*)(frag + (size_t)(b * kNC + ch + 1) * kChunkH),
                     tid);

    // serial (tiny) stabilizer recurrence: F_s, m_s
    if (tid == 0) {
      float F = 0.f, mp = mcar[0];
      for (int s = 0; s < kL; ++s) {
        F += fbs[s];
        Fc[s] = F;
        mp = fmaxf(fbs[s] + mp, ibs[s]);
        msv[s] = mp;
      }
    }
    __syncthreads();

    // W[s][j] = exp(F_s - F_j + ib_j - m_s) for j<=s (<=1); scatter into WA.
    for (int i = tid; i < kL * kL; i += 256) {
      int s = i >> 5, j = i & (kL - 1);
      float wv = (j <= s) ? __expf(Fc[s] - Fc[j] + ibs[j] - msv[s]) : 0.f;
      Wm[i] = wv;
      WA[a_scatter_idx(s, j)] = (_Float16)wv;
    }
    if (tid < kL) dec[tid] = __expf(Fc[tid] + mcar[0] - msv[tid]);
    __syncthreads();

    // Phase A (waves 0-3): Shat = W .* (Q K^T), scattered into A-frag layout.
    // Phase A (waves 4-7): prebuild (w_L o V)^T A-fragments from VsB.
    if (wave < 4) {
      int si = wave >> 1, sj = wave & 1;
      v8f acc = {};
#pragma unroll
      for (int kk = 0; kk < 4; ++kk) {
        v16h a  = frag_ld(QsA + (si * 4 + kk) * kFragH, lane);
        v16h bf = frag_ld(KsT + (kk * 2 + sj) * kFragH, lane);
        acc = wmma_f16(a, bf, acc);
      }
#pragma unroll
      for (int r = 0; r < 8; ++r) {
        int s = si * 16 + r + hi * 8;
        int j = sj * 16 + m16;
        SsA[a_scatter_idx(s, j)] = (_Float16)(Wm[s * kL + j] * acc[r]);
      }
    } else {
      const float* wL = Wm + (kL - 1) * kL;
      for (int i2 = tid - 128; i2 < kD * kL; i2 += 128) {
        int irow = i2 >> 5, j = i2 & 31;             // state row i, chunk step j
        float vvv = (float)VsB[b_frag_idx(j, irow)];
        VA[a_scatter_idx(irow, j)] = (_Float16)(wL[j] * vvv);
      }
    }
    __syncthreads();

    // Phase B: SV = Shat@V, QC = Q@C0^T, NM = W@K   (each [L,D], 16 tiles each)
    for (int t = wave; t < 48; t += 8) {
      int kind = t >> 4;
      int tt = t & 15;
      int mi = tt >> 3, ni = tt & 7;
      v8f acc = {};
      if (kind == 0) {                 // Shat [L,L] @ V [L,D], K=32 single step
        v16h a  = frag_ld(SsA + mi * kFragH, lane);
        v16h bf = frag_ld(VsB + ni * kFragH, lane);
        acc = wmma_f16(a, bf, acc);
      } else if (kind == 1) {          // Q [L,D] @ C0^T via C0F frags
#pragma unroll
        for (int kk = 0; kk < 4; ++kk) {
          v16h a  = frag_ld(QsA + (mi * 4 + kk) * kFragH, lane);
          v16h bf = frag_ld(C0F + (kk * 8 + ni) * kFragH, lane);
          acc = wmma_f16(a, bf, acc);
        }
      } else {                         // W [L,L] @ K [L,D]
        v16h a  = frag_ld(WA + mi * kFragH, lane);
        v16h bf = frag_ld(KsB + ni * kFragH, lane);
        acc = wmma_f16(a, bf, acc);
      }
      float* dst = (kind == 0) ? SVb : (kind == 1) ? QCb : NMb;
#pragma unroll
      for (int r = 0; r < 8; ++r)
        dst[(mi * 16 + r + hi * 8) * kD + ni * 16 + m16] = acc[r];
    }
    __syncthreads();

    // denominators: den[s] = max(1, sum_d |(dec_s*n0 + NM[s]) * q_s|); new n
    for (int qq = 0; qq < 4; ++qq) {
      int s = wave * 4 + qq;
      float part = 0.f;
      for (int d0c = lane; d0c < kD; d0c += 32) {
        float nv = dec[s] * n0[d0c] + NMb[s * kD + d0c];
        part += fabsf(nv * (float)QsA[a_frag_idx(s, d0c)]);
      }
#pragma unroll
      for (int off = 16; off > 0; off >>= 1) part += __shfl_xor(part, off, 32);
      if (lane == 0) den[s] = fmaxf(part, 1.0f);
    }
    if (tid < kD) nnew[tid] = dec[kL - 1] * n0[tid] + NMb[(kL - 1) * kD + tid];
    __syncthreads();

    // h output:  h = o * (dec_s * QC + SV) / den_s
    for (int i = tid; i < kL * kD; i += 256) {
      int s = i >> 7, d0c = i & (kD - 1);
      float hv = og[rowbase + i] * (dec[s] * QCb[i] + SVb[i]) / den[s];
      outH[rowbase + i] = hv;
      if (t0 + s == kT - 1) outHlast[b * kD + d0c] = hv;
    }

    // Phase C: C0 = dec_L*C0 + (w_L o V)^T @ K  (8x8 tiles of [D,D], K=32);
    // maintain the f16 shadow C0F (QC B-layout) for next chunk.
    {
      const float dL = dec[kL - 1];
      for (int t = wave; t < 64; t += 8) {
        int mi = t >> 3, ni = t & 7;
        v16h a  = frag_ld(VA + mi * kFragH, lane);
        v16h bf = frag_ld(KsB + ni * kFragH, lane);
        v8f acc = {};
        acc = wmma_f16(a, bf, acc);
#pragma unroll
        for (int r = 0; r < 8; ++r) {
          int ii = mi * 16 + r + hi * 8;
          int dd = ni * 16 + m16;
          float nv = dL * C0[ii * kD + dd] + acc[r];
          C0[ii * kD + dd] = nv;
          C0F[c0f_idx(ii, dd)] = (_Float16)nv;
        }
      }
    }
    __syncthreads();
    if (tid < kD) n0[tid] = nnew[tid];
    if (tid == 0) mcar[0] = msv[kL - 1];
    __syncthreads();
  }

  // final state outputs
  for (int i = tid; i < kD * kD; i += 256) outC[(size_t)b * kD * kD + i] = C0[i];
  if (tid < kD) {
    outN[b * kD + tid] = n0[tid];
    outM[b * kD + tid] = mcar[0];
  }
}

extern "C" void kernel_launch(void* const* d_in, const int* in_sizes, int n_in,
                              void* d_out, int out_size, void* d_ws, size_t ws_size,
                              hipStream_t stream) {
  (void)in_sizes; (void)n_in; (void)out_size; (void)ws_size;
  const float* x  = (const float*)d_in[0];
  const float* Wi = (const float*)d_in[1];
  const float* bi = (const float*)d_in[2];
  const float* Wf = (const float*)d_in[3];
  const float* bf = (const float*)d_in[4];
  const float* Wo = (const float*)d_in[5];
  const float* bo = (const float*)d_in[6];
  const float* Wq = (const float*)d_in[7];
  const float* bq = (const float*)d_in[8];
  const float* Wk = (const float*)d_in[9];
  const float* bk = (const float*)d_in[10];
  const float* Wv = (const float*)d_in[11];
  const float* bv = (const float*)d_in[12];
  float* out = (float*)d_out;

  const size_t rows = (size_t)kB * kT;
  _Float16* wsFrag = (_Float16*)d_ws;                       // B*NC*32KB = 16MB
  float* wsO = (float*)(wsFrag + (size_t)kB * kNC * kChunkH);
  float* wsI = wsO + rows * kD;
  float* wsF = wsI + rows;

  mlstm_proj_kernel<<<(int)(rows / 16), 256, 0, stream>>>(
      x, Wq, bq, Wk, bk, Wv, bv, Wo, bo, wsFrag, wsO);
  mlstm_gate_kernel<<<(int)((rows + 255) / 256), 256, 0, stream>>>(
      x, Wi, bi, Wf, bf, wsI, wsF, (int)rows);

  // LDS: f16 region (C0F 32 + 2x32 QKV + VA 8 + SsA 2 + WA 2 frags) + fp32.
  size_t lds = (32u + 64u + 8u + 2u + 2u) * kFragH * sizeof(_Float16);
  size_t fp32_words = (size_t)kD * kD + 3u * kL * kD + (size_t)kL * kL +
                      2u * kD + 6u * kL + 1;
  lds += fp32_words * sizeof(float);
  lds = (lds + 63) & ~(size_t)63;
  mlstm_scan_kernel<<<kB, 256, lds, stream>>>(wsFrag, wsO, wsI, wsF, out);
}
